// SelfAttention_43490838839638
// MI455X (gfx1250) — compile-verified, matrix operand order checked by reading, not run on hardware
//
#include <hip/hip_runtime.h>
#include <hip/hip_bf16.h>
#include <math.h>

// ---------------------------------------------------------------------------
// Self-attention (B=8, C=256, H=W=64) for MI455X / gfx1250, wave32 + WMMA.
//   Phase 0: convert wq/wk/wv/wo fp32 -> f16 (workspace)
//   Phase 1: Q/K/V 1x1 projections via v_wmma_f32_16x16x32_f16
//   Phase 2: flash-attention; K/V tiles double-buffered into LDS with
//            global_load_async_to_lds_b128 (ASYNCcnt), WMMA from LDS
//   Phase 3: output projection via WMMA + fused bias/residual (b128 I/O)
// ---------------------------------------------------------------------------

typedef __attribute__((ext_vector_type(16))) _Float16 v16h;
typedef __attribute__((ext_vector_type(8)))  _Float16 v8h;
typedef __attribute__((ext_vector_type(8)))  float    v8f;
typedef __attribute__((ext_vector_type(4)))  float    v4f;

#define BATCH 8
#define CH    256
#define DK    128
#define NPIX  4096   // 64*64

// ---------------------------------------------------------------------------
// Fragment loader. CDNA5 16-bit A (16x32) and B (32x16) operands share the
// same per-lane K layout (ISA 7.12.2):
//   lanes 0-15 : elements 0..7 = K k0+0..7,  elements 8..15 = K k0+16..23
//   lanes 16-31: elements 0..7 = K k0+8..15, elements 8..15 = K k0+24..31
// With row-major (contiguous-K) storage this is exactly two b128 loads.
// ---------------------------------------------------------------------------
__device__ __forceinline__ v16h load_frag_h(const _Float16* __restrict__ row_ptr,
                                            int k0, int hi) {
  const v8h lo = *reinterpret_cast<const v8h*>(row_ptr + k0 + hi * 8);
  const v8h hih = *reinterpret_cast<const v8h*>(row_ptr + k0 + 16 + hi * 8);
  v16h r;
#pragma unroll
  for (int i = 0; i < 8; ++i) { r[i] = lo[i]; r[i + 8] = hih[i]; }
  return r;
}

__device__ __forceinline__ v8f wmma_f16(v16h a, v16h b, v8f c) {
  return __builtin_amdgcn_wmma_f32_16x16x32_f16(
      /*neg_a=*/false, a, /*neg_b=*/false, b,
      /*c_mod=*/(short)0, c, /*reuse_a=*/false, /*reuse_b=*/false);
}

// --- gfx1250 async copy: 16B global -> LDS per lane, tracked by ASYNCcnt ---
__device__ __forceinline__ void async_copy16(void* lds_dst, const void* gsrc) {
  const unsigned l = (unsigned)(uintptr_t)lds_dst;  // LDS byte addr = low 32b
  asm volatile("global_load_async_to_lds_b128 %0, %1, off"
               :: "v"(l), "v"(gsrc) : "memory");
}
__device__ __forceinline__ void wait_async0() {
  asm volatile("s_wait_asynccnt 0" ::: "memory");
}

// ---------------------------------------------------------------------------
// Phase 0: weights fp32 -> f16 (wq, wk, wv: [DK][CH]; wo: [CH][DK])
// ---------------------------------------------------------------------------
__global__ __launch_bounds__(256) void convert_w_kernel(
    const float* __restrict__ wq, const float* __restrict__ wk,
    const float* __restrict__ wv, const float* __restrict__ wo,
    _Float16* __restrict__ wqh, _Float16* __restrict__ wkh,
    _Float16* __restrict__ wvh, _Float16* __restrict__ woh) {
  int i = blockIdx.x * 256 + threadIdx.x;
  if (i < DK * CH) {
    wqh[i] = (_Float16)wq[i];
    wkh[i] = (_Float16)wk[i];
    wvh[i] = (_Float16)wv[i];
    woh[i] = (_Float16)wo[i];
  }
}

// ---------------------------------------------------------------------------
// Phase 1: QKV projections.
// Q[b][n][d] = sum_c x[b][c][n] * wq[d][c] + bq[d]   (A = x^T tile, B = wq row)
// Block = 128 threads (4 waves), 64 pixels of one batch.
// Q,K stored [B][N][128] f16 (K-major for fragments); V stored [B][128][N].
// ---------------------------------------------------------------------------
__device__ __forceinline__ void proj_one(const v16h afr[8],
                                         const _Float16* __restrict__ wh,
                                         const float* __restrict__ bias,
                                         _Float16* __restrict__ out, int b,
                                         int n_base, int hi, int l16,
                                         bool v_layout) {
#pragma unroll
  for (int dt = 0; dt < 8; ++dt) {
    const int d = dt * 16 + l16;
    v8f acc = {};
#pragma unroll
    for (int kc = 0; kc < 8; ++kc) {             // C = 256 = 8 * 32
      v16h bf = load_frag_h(wh + (size_t)d * CH, kc * 32, hi);
      acc = wmma_f16(afr[kc], bf, acc);
    }
    const float bb = bias[d];
#pragma unroll
    for (int i = 0; i < 8; ++i) {                // C-layout: VGPR i = row
      const int n = n_base + i + hi * 8;
      const float v = acc[i] + bb;
      if (!v_layout)
        out[((size_t)b * NPIX + n) * DK + d] = (_Float16)v;   // [B][N][DK]
      else
        out[((size_t)b * DK + d) * NPIX + n] = (_Float16)v;   // [B][DK][N]
    }
  }
}

__global__ __launch_bounds__(128) void qkv_proj_kernel(
    const float* __restrict__ x, const _Float16* __restrict__ wqh,
    const _Float16* __restrict__ wkh, const _Float16* __restrict__ wvh,
    const float* __restrict__ bq, const float* __restrict__ bk,
    const float* __restrict__ bv, _Float16* __restrict__ Qh,
    _Float16* __restrict__ Kh, _Float16* __restrict__ Vh) {
  __shared__ _Float16 xT[64 * CH];               // [n_local][c], 32 KB

  const int blk = blockIdx.x;                    // 8 * 64 blocks
  const int b = blk >> 6;
  const int n0 = (blk & 63) * 64;
  const int tid = threadIdx.x;
  const int nl = tid & 63;

  const float* xb = x + (size_t)b * CH * NPIX;
  for (int c = tid >> 6; c < CH; c += 2) {
    const float v = xb[(size_t)c * NPIX + n0 + nl];
    xT[nl * CH + c] = (_Float16)v;
  }
  __syncthreads();

  const int wid = tid >> 5, lane = tid & 31;
  const int hi = lane >> 4, l16 = lane & 15;
  const int row = wid * 16 + l16;

  v16h afr[8];                                   // hoist all 8 K-chunks of A
#pragma unroll
  for (int kc = 0; kc < 8; ++kc)
    afr[kc] = load_frag_h(&xT[row * CH], kc * 32, hi);

  const int n_base = n0 + wid * 16;
  proj_one(afr, wqh, bq, Qh, b, n_base, hi, l16, false);
  proj_one(afr, wkh, bk, Kh, b, n_base, hi, l16, false);
  proj_one(afr, wvh, bv, Vh, b, n_base, hi, l16, true);
}

// ---------------------------------------------------------------------------
// Phase 2: flash attention. 8 waves/block, each wave owns a 16-query tile;
// all 8 tiles share one double-buffered K/V stream staged into LDS with
// async-to-LDS copies (8x reuse vs per-wave streaming). Online softmax with
// width-16 shuffles; P reshaped C-layout -> A-fragment via per-wave LDS.
// Output O stored f16 [B][N][DK] (fragment-ready for phase 3).
// ---------------------------------------------------------------------------
__global__ __launch_bounds__(256) void flash_attn_kernel(
    const _Float16* __restrict__ Qh, const _Float16* __restrict__ Kh,
    const _Float16* __restrict__ Vh, _Float16* __restrict__ Ofh) {
  __shared__ _Float16 Kt[2][32 * DK];            // [key][d]   8 KB per buffer
  __shared__ _Float16 Vt[2][DK * 32];            // [d][key]   8 KB per buffer
  __shared__ _Float16 Pst[8][16 * 32];           // per-wave P staging, 8 KB

  const int tid = threadIdx.x, wid = tid >> 5, lane = tid & 31;
  const int hi = lane >> 4, l16 = lane & 15;
  const int t = blockIdx.x * 8 + wid;            // 256 blocks * 8 waves = 2048
  const int b = t >> 8;                          // uniform per block (256%8==0)
  const int n_base = (t & 255) * 16;

  const float scale_log2 = 1.44269504088896f * 0.0883883476483184f;

  const _Float16* kbase = Kh + (size_t)b * NPIX * DK;
  const _Float16* vbase = Vh + (size_t)b * DK * NPIX;

  // Stage one 32-key K/V tile into LDS buffer `buf` (async, per-thread 16B).
  auto stage = [&](int buf, int m0) {
    const _Float16* gk = kbase + (size_t)m0 * DK;      // 8 KB contiguous
#pragma unroll
    for (int i = 0; i < 2; ++i) {
      const int c = tid + i * 256;                     // 512 x 16B chunks
      async_copy16(&Kt[buf][c * 8], gk + (size_t)c * 8);
    }
#pragma unroll
    for (int i = 0; i < 2; ++i) {
      const int c = tid + i * 256;                     // d = c/4, 4 x 16B/row
      const int d = c >> 2, sub = c & 3;
      async_copy16(&Vt[buf][d * 32 + sub * 8],
                   vbase + (size_t)d * NPIX + m0 + sub * 8);
    }
  };

  const _Float16* qrow = Qh + ((size_t)b * NPIX + n_base + l16) * DK;
  v16h qfr[4];
#pragma unroll
  for (int kc = 0; kc < 4; ++kc) qfr[kc] = load_frag_h(qrow, kc * 32, hi);

  v8f o[8];
#pragma unroll
  for (int dt = 0; dt < 8; ++dt) o[dt] = v8f{};
  float mrow[8], lrow[8];
#pragma unroll
  for (int i = 0; i < 8; ++i) { mrow[i] = -INFINITY; lrow[i] = 0.0f; }

  stage(0, 0);
  wait_async0();
  __syncthreads();

  for (int it = 0; it < NPIX / 32; ++it) {
    const int cur = it & 1;
    if (it + 1 < NPIX / 32) stage(cur ^ 1, (it + 1) * 32);  // overlap fill

    // ---- S = Q * K^T for 32 keys (two 16-col tiles), operands from LDS ----
    v8f s0 = {}, s1 = {};
    const _Float16* krow0 = &Kt[cur][l16 * DK];
    const _Float16* krow1 = &Kt[cur][(16 + l16) * DK];
#pragma unroll
    for (int kc = 0; kc < 4; ++kc) {
      s0 = wmma_f16(qfr[kc], load_frag_h(krow0, kc * 32, hi), s0);
      s1 = wmma_f16(qfr[kc], load_frag_h(krow1, kc * 32, hi), s1);
    }

    // ---- online softmax (VGPR index i = row within half-wave) ----
#pragma unroll
    for (int i = 0; i < 8; ++i) {
      float mx = fmaxf(s0[i], s1[i]);
      mx = fmaxf(mx, __shfl_xor(mx, 1, 16));
      mx = fmaxf(mx, __shfl_xor(mx, 2, 16));
      mx = fmaxf(mx, __shfl_xor(mx, 4, 16));
      mx = fmaxf(mx, __shfl_xor(mx, 8, 16));
      const float mnew = fmaxf(mrow[i], mx);
      const float alpha = __builtin_exp2f((mrow[i] - mnew) * scale_log2);
      const float p0 = __builtin_exp2f((s0[i] - mnew) * scale_log2);
      const float p1 = __builtin_exp2f((s1[i] - mnew) * scale_log2);
      float rs = p0 + p1;
      rs += __shfl_xor(rs, 1, 16);
      rs += __shfl_xor(rs, 2, 16);
      rs += __shfl_xor(rs, 4, 16);
      rs += __shfl_xor(rs, 8, 16);
      lrow[i] = lrow[i] * alpha + rs;
      mrow[i] = mnew;
#pragma unroll
      for (int dt = 0; dt < 8; ++dt) o[dt][i] *= alpha;
      const int r = i + hi * 8;
      Pst[wid][r * 32 + l16] = (_Float16)p0;
      Pst[wid][r * 32 + 16 + l16] = (_Float16)p1;
    }
    asm volatile("s_wait_dscnt 0" ::: "memory");   // P staging visible (wave)

    // ---- O += P(16x32) * V^T(32x16) per 16-wide d tile ----
    v16h pa = load_frag_h(&Pst[wid][l16 * 32], 0, hi);
#pragma unroll
    for (int dt = 0; dt < 8; ++dt) {
      const int d = dt * 16 + l16;
      v16h bf = load_frag_h(&Vt[cur][d * 32], 0, hi);
      o[dt] = wmma_f16(pa, bf, o[dt]);
    }

    wait_async0();        // next buffer fully in LDS (this wave's copies)
    __syncthreads();      // all waves done filling + done reading `cur`
  }

  // ---- epilogue: O /= l, store f16 [B][N][DK] ----
  float invl[8];
#pragma unroll
  for (int i = 0; i < 8; ++i) invl[i] = 1.0f / lrow[i];
#pragma unroll
  for (int dt = 0; dt < 8; ++dt) {
    const int d = dt * 16 + l16;
#pragma unroll
    for (int i = 0; i < 8; ++i) {
      const int n = n_base + i + hi * 8;
      Ofh[((size_t)b * NPIX + n) * DK + d] = (_Float16)(o[dt][i] * invl[i]);
    }
  }
}

// ---------------------------------------------------------------------------
// Phase 3: y = x + wo @ O + bo via WMMA.
// Each wave owns a 16-pixel tile: A = O tile (16n x 128d, same fragments as
// Q), loop 16 c-tiles: B = wo rows (contiguous d), 4 WMMAs each. C/D layout
// gives each lane 8 consecutive n at fixed c -> b128 residual load/store.
// ---------------------------------------------------------------------------
__global__ __launch_bounds__(256) void out_proj_kernel(
    const float* __restrict__ x, const _Float16* __restrict__ woh,
    const float* __restrict__ bo, const _Float16* __restrict__ Ofh,
    float* __restrict__ y) {
  const int tid = threadIdx.x, wid = tid >> 5, lane = tid & 31;
  const int hi = lane >> 4, l16 = lane & 15;
  const int t = blockIdx.x * 8 + wid;            // 2048 pixel tiles
  const int b = t >> 8;
  const int n_base = (t & 255) * 16;

  const _Float16* orow = Ofh + ((size_t)b * NPIX + n_base + l16) * DK;
  v16h afr[4];
#pragma unroll
  for (int kc = 0; kc < 4; ++kc) afr[kc] = load_frag_h(orow, kc * 32, hi);

  const int n0 = n_base + hi * 8;                // lane's 8 consecutive pixels
#pragma unroll
  for (int ct = 0; ct < 16; ++ct) {
    const int c = ct * 16 + l16;                 // lane's output channel
    v8f acc = {};
#pragma unroll
    for (int kc = 0; kc < 4; ++kc) {             // d_k = 128 = 4 * 32
      v16h bf = load_frag_h(woh + (size_t)c * DK, kc * 32, hi);
      acc = wmma_f16(afr[kc], bf, acc);
    }
    const float bb = bo[c];
    const float* xp = x + ((size_t)b * CH + c) * NPIX + n0;
    float*       yp = y + ((size_t)b * CH + c) * NPIX + n0;
    const v4f x0 = *reinterpret_cast<const v4f*>(xp);
    const v4f x1 = *reinterpret_cast<const v4f*>(xp + 4);
    v4f y0, y1;
#pragma unroll
    for (int i = 0; i < 4; ++i) {
      y0[i] = x0[i] + acc[i] + bb;
      y1[i] = x1[i] + acc[i + 4] + bb;
    }
    *reinterpret_cast<v4f*>(yp) = y0;
    *reinterpret_cast<v4f*>(yp + 4) = y1;
  }
}

// ---------------------------------------------------------------------------
// Launch. Workspace: wqh/wkh/wvh/woh (4*64KB) | Qh,Kh [B][N][DK] f16 (8MB
// each) | Vh [B][DK][N] f16 (8MB) | Ofh [B][N][DK] f16 (8MB)  => ~32.3 MB
// ---------------------------------------------------------------------------
extern "C" void kernel_launch(void* const* d_in, const int* in_sizes, int n_in,
                              void* d_out, int out_size, void* d_ws,
                              size_t ws_size, hipStream_t stream) {
  const float* x  = (const float*)d_in[0];
  const float* wq = (const float*)d_in[1];
  const float* bq = (const float*)d_in[2];
  const float* wk = (const float*)d_in[3];
  const float* bk = (const float*)d_in[4];
  const float* wv = (const float*)d_in[5];
  const float* bv = (const float*)d_in[6];
  const float* wo = (const float*)d_in[7];
  const float* bo = (const float*)d_in[8];
  float* y = (float*)d_out;

  char* ws = (char*)d_ws;
  const size_t W_BYTES = (size_t)DK * CH * sizeof(_Float16);
  const size_t QKV_BYTES = (size_t)BATCH * NPIX * DK * sizeof(_Float16);
  _Float16* wqh = (_Float16*)(ws);
  _Float16* wkh = (_Float16*)(ws + W_BYTES);
  _Float16* wvh = (_Float16*)(ws + 2 * W_BYTES);
  _Float16* woh = (_Float16*)(ws + 3 * W_BYTES);
  _Float16* Qh  = (_Float16*)(ws + 4 * W_BYTES);
  _Float16* Kh  = (_Float16*)(ws + 4 * W_BYTES + QKV_BYTES);
  _Float16* Vh  = (_Float16*)(ws + 4 * W_BYTES + 2 * QKV_BYTES);
  _Float16* Ofh = (_Float16*)(ws + 4 * W_BYTES + 3 * QKV_BYTES);

  convert_w_kernel<<<(DK * CH + 255) / 256, 256, 0, stream>>>(
      wq, wk, wv, wo, wqh, wkh, wvh, woh);
  qkv_proj_kernel<<<BATCH * (NPIX / 64), 128, 0, stream>>>(
      x, wqh, wkh, wvh, bq, bk, bv, Qh, Kh, Vh);
  flash_attn_kernel<<<(BATCH * (NPIX / 16)) / 8, 256, 0, stream>>>(Qh, Kh, Vh,
                                                                   Ofh);
  out_proj_kernel<<<(BATCH * (NPIX / 16)) / 8, 256, 0, stream>>>(x, woh, bo,
                                                                 Ofh, y);
}
